// SRAttentionBlock_57337813401887
// MI455X (gfx1250) — compile-verified, matrix operand order checked by reading, not run on hardware
//
#include <hip/hip_runtime.h>
#include <hip/hip_bf16.h>
#include <math.h>

// ---------------- problem constants ----------------
#define BB 4
#define CC 64
#define MM 48
#define VV (48*48*48)      // 110592
#define AA 64
#define EE 128

typedef __attribute__((ext_vector_type(16))) _Float16 v16h;
typedef __attribute__((ext_vector_type(8)))  _Float16 v8h;
typedef __attribute__((ext_vector_type(8)))  float    v8f;

static __device__ __forceinline__ v16h cat8(v8h lo, v8h hi) {
    return __builtin_shufflevector(lo, hi, 0,1,2,3,4,5,6,7,8,9,10,11,12,13,14,15);
}
static __device__ __forceinline__ v8f vzero8() {
    v8f z = {0.f,0.f,0.f,0.f,0.f,0.f,0.f,0.f};
    return z;
}
static __device__ __forceinline__ int wrap48(int q) {
    q += (q < 0) ? 48 : 0;
    q -= (q >= 48) ? 48 : 0;
    return q;
}

// ---------------- workspace layout (in _Float16 units) ----------------
// vmat    : [B][A][C]                      = 16384
// vpack   : [B][2kt*4nt][32][16]           = 16384
// wout    : [4kt*4nt][32][16]              = 8192
// wconv   : [27tap*2kt*4nt][32][16]        = 110592
// hin     : [B][V][128]                    = 56623104
// h       : [B][V][64]                     = 28311552
#define OFF_VMAT   ((size_t)0)
#define OFF_VPACK  ((size_t)16384)
#define OFF_WOUT   ((size_t)32768)
#define OFF_WCONV  ((size_t)40960)
#define OFF_HIN    ((size_t)151552)
#define OFF_H      ((size_t)(151552 + 56623104))

// =====================================================================
// 1) v = mol_embed @ v_w.T + v_b  ->  f16 vmat[b][a][c]
// =====================================================================
__global__ void prep_v_kernel(const float* __restrict__ me,
                              const float* __restrict__ v_w,
                              const float* __restrict__ v_b,
                              _Float16* __restrict__ vmat) {
    int tid = blockIdx.x * blockDim.x + threadIdx.x;
    if (tid >= BB * AA * CC) return;
    int c = tid & 63;
    int a = (tid >> 6) & 63;
    int b = tid >> 12;
    const float* mrow = me + ((size_t)(b * AA + a)) * EE;
    const float* wrow = v_w + (size_t)c * EE;
    float s = v_b[c];
    #pragma unroll 8
    for (int e = 0; e < EE; ++e) s += mrow[e] * wrow[e];
    vmat[((size_t)(b * AA + a)) * CC + c] = (_Float16)s;
}

// =====================================================================
// 2) pack B matrices into per-lane WMMA fragment layout.
//    For V_WMMA_F32_16X16X32_F16 B-frag: lane l holds column n = l&15,
//    K values ks0..ks0+7 (halves 0..7) and ks0+16..ks0+23 (halves 8..15),
//    ks0 = kt*32 + (l<16 ? 0 : 8).  Each lane's 16 halves stored contiguous.
// =====================================================================
__global__ void pack_all_kernel(const float* __restrict__ out_w,
                                const float* __restrict__ conv_w,
                                const _Float16* __restrict__ vmat,
                                _Float16* __restrict__ wout,
                                _Float16* __restrict__ wconv,
                                _Float16* __restrict__ vpack) {
    int tid = blockIdx.x * blockDim.x + threadIdx.x;
    int lane = tid & 31;
    int base = (lane < 16) ? 0 : 8;
    int n0 = lane & 15;
    if (tid < 512) {                       // ---- out_w [N=64][K=128] ----
        int frag = tid >> 5;               // kt*4 + nt
        int kt = frag >> 2, nt = frag & 3;
        int n = nt * 16 + n0;
        int ks0 = kt * 32 + base;
        _Float16* dst = wout + ((size_t)frag * 32 + lane) * 16;
        #pragma unroll
        for (int j = 0; j < 8; ++j) {
            dst[j]     = (_Float16)out_w[(size_t)n * 128 + (ks0 + j)];
            dst[8 + j] = (_Float16)out_w[(size_t)n * 128 + (ks0 + 16 + j)];
        }
    } else if (tid < 512 + 216 * 32) {     // ---- conv_w [O][I][3][3][3] ----
        int t2 = tid - 512;
        int f = t2 >> 5;                   // (tap*2+kt)*4 + nt
        int tap = f >> 3;
        int kt = (f >> 2) & 1, nt = f & 3;
        int dx = tap / 9, dy = (tap / 3) % 3, dz = tap % 3;
        int n = nt * 16 + n0;
        int ks0 = kt * 32 + base;
        _Float16* dst = wconv + ((size_t)f * 32 + lane) * 16;
        #pragma unroll
        for (int j = 0; j < 8; ++j) {
            int k0 = ks0 + j, k1 = ks0 + 16 + j;
            dst[j]     = (_Float16)conv_w[((((size_t)n * 64 + k0) * 3 + dx) * 3 + dy) * 3 + dz];
            dst[8 + j] = (_Float16)conv_w[((((size_t)n * 64 + k1) * 3 + dx) * 3 + dy) * 3 + dz];
        }
    } else if (tid < 512 + 216 * 32 + BB * 256) {  // ---- vmat [K=A][N=C] ----
        int t2 = tid - (512 + 216 * 32);
        int b = t2 >> 8;
        int frag = (t2 >> 5) & 7;          // kt*4 + nt, kt in 0..1
        int kt = frag >> 2, nt = frag & 3;
        int n = nt * 16 + n0;
        int ks0 = kt * 32 + base;
        _Float16* dst = vpack + (((size_t)b * 8 + frag) * 32 + lane) * 16;
        #pragma unroll
        for (int j = 0; j < 8; ++j) {
            dst[j]     = vmat[((size_t)b * AA + (ks0 + j)) * CC + n];
            dst[8 + j] = vmat[((size_t)b * AA + (ks0 + 16 + j)) * CC + n];
        }
    }
}

// =====================================================================
// 3) transpose x[b][c][v] (f32) -> hin[b][v][c] (f16), c < 64
// =====================================================================
__global__ void prep_xt_kernel(const float* __restrict__ x,
                               _Float16* __restrict__ hin) {
    int tid = blockIdx.x * blockDim.x + threadIdx.x;   // one thread = 4 v's
    const int VQ = VV / 4;
    int b = tid / (CC * VQ);
    int c = (tid / VQ) % CC;
    int v4 = (tid % VQ) * 4;
    const float4 xv = *(const float4*)(x + ((size_t)(b * CC + c)) * VV + v4);
    _Float16* dst = hin + ((size_t)b * VV + v4) * 128 + c;
    dst[0]       = (_Float16)xv.x;
    dst[128]     = (_Float16)xv.y;
    dst[256]     = (_Float16)xv.z;
    dst[384]     = (_Float16)xv.w;
}

// =====================================================================
// 4) attention: distances -> masked softmax -> att[16x64] @ v[64x64]
//    block = 128 threads = 4 waves; wave handles one 16-row M-tile.
// =====================================================================
__global__ void attn_kernel(const float* __restrict__ pos,
                            const float* __restrict__ origin,
                            const float* __restrict__ lattice,
                            const int* __restrict__ batch_nodes,
                            const _Float16* __restrict__ vpack,
                            _Float16* __restrict__ hin) {
    __shared__ __align__(16) _Float16 abuf[4][16][72];   // +8 pad halves

    const int lane = threadIdx.x & 31;
    const int wv   = threadIdx.x >> 5;
    const int b    = blockIdx.y;
    const int mtile = blockIdx.x * 4 + wv;
    const int v0 = mtile * 16;
    const int r = lane & 15;
    const int base = (lane < 16) ? 0 : 8;
    const int a0 = (lane >> 4) * 32;      // this lane's atom half

    const int iz0 = v0 % 48;
    const int iy  = (v0 / 48) % 48;
    const int ix  = v0 / 2304;

    const float o0 = origin[b * 3 + 0], o1 = origin[b * 3 + 1], o2 = origin[b * 3 + 2];
    const float l00 = lattice[b * 9 + 0], l11 = lattice[b * 9 + 4], l22 = lattice[b * 9 + 8];
    const float inv47 = 1.0f / 47.0f;
    const float gx = o0 + l00 * ((float)ix * inv47);
    const float gy = o1 + l11 * ((float)iy * inv47);
    const float gz = o2 + l22 * ((float)(iz0 + r) * inv47);
    const int bn = batch_nodes[b];

    float w[32];
    float mx = -INFINITY;
    #pragma unroll
    for (int i = 0; i < 32; ++i) {
        int a = a0 + i;
        const float* p = pos + ((size_t)(b * AA + a)) * 3;
        float dxv = gx - p[0], dyv = gy - p[1], dzv = gz - p[2];
        float d = sqrtf(fmaxf(dxv*dxv + dyv*dyv + dzv*dzv, 1e-12f));
        float wi = (a < bn) ? -d : -INFINITY;
        w[i] = wi;
        mx = fmaxf(mx, wi);
    }
    mx = fmaxf(mx, __shfl_xor(mx, 16, 32));   // combine the two half-row lanes
    float s = 0.f;
    #pragma unroll
    for (int i = 0; i < 32; ++i) { float e = __expf(w[i] - mx); w[i] = e; s += e; }
    s += __shfl_xor(s, 16, 32);
    const float inv = 1.0f / s;
    #pragma unroll
    for (int i = 0; i < 32; ++i) abuf[wv][r][a0 + i] = (_Float16)(w[i] * inv);
    __syncthreads();

    v8f acc[4] = { vzero8(), vzero8(), vzero8(), vzero8() };
    const _Float16* arow = &abuf[wv][r][0];
    #pragma unroll
    for (int kt = 0; kt < 2; ++kt) {
        int ks = kt * 32 + base;
        v8h alo = *(const v8h*)(arow + ks);
        v8h ahi = *(const v8h*)(arow + ks + 16);
        v16h afrag = cat8(alo, ahi);
        #pragma unroll
        for (int nt = 0; nt < 4; ++nt) {
            v16h bfrag = *(const v16h*)(vpack + (((size_t)b * 8 + kt * 4 + nt) * 32 + lane) * 16);
            acc[nt] = __builtin_amdgcn_wmma_f32_16x16x32_f16(
                false, afrag, false, bfrag, (short)0, acc[nt], false, false);
        }
    }
    // D layout: lane holds column n = lane&15, rows j + (lane<16?0:8)
    const int n0 = lane & 15;
    const int rb = (lane < 16) ? 0 : 8;
    #pragma unroll
    for (int nt = 0; nt < 4; ++nt) {
        #pragma unroll
        for (int j = 0; j < 8; ++j) {
            hin[((size_t)(b * VV + v0 + rb + j)) * 128 + 64 + nt * 16 + n0] =
                (_Float16)acc[nt][j];
        }
    }
}

// =====================================================================
// 5) h = relu(hin[V,128] @ out_w.T + out_b) -> h f16 [b][v][64]
//    block = 256 threads = 8 waves; wave = one 32-row M-tile (two 16-row
//    WMMA tiles sharing every B fragment); 32 WMMAs/wave.
// =====================================================================
__global__ void out_gemm_kernel(const _Float16* __restrict__ hin,
                                const _Float16* __restrict__ wout,
                                const float* __restrict__ out_b,
                                _Float16* __restrict__ h) {
    const int lane = threadIdx.x & 31;
    const int wv   = threadIdx.x >> 5;
    const int b    = blockIdx.y;
    const int mtile = blockIdx.x * 8 + wv;     // 32-row tiles
    const int v0 = mtile * 32;
    const int r = lane & 15;
    const int base = (lane < 16) ? 0 : 8;

    v8f acc0[4] = { vzero8(), vzero8(), vzero8(), vzero8() };
    v8f acc1[4] = { vzero8(), vzero8(), vzero8(), vzero8() };
    const _Float16* arow0 = hin + ((size_t)(b * VV + v0 + r)) * 128;
    const _Float16* arow1 = arow0 + (size_t)16 * 128;
    #pragma unroll
    for (int kt = 0; kt < 4; ++kt) {
        int ks = kt * 32 + base;
        v16h a0 = cat8(*(const v8h*)(arow0 + ks), *(const v8h*)(arow0 + ks + 16));
        v16h a1 = cat8(*(const v8h*)(arow1 + ks), *(const v8h*)(arow1 + ks + 16));
        #pragma unroll
        for (int nt = 0; nt < 4; ++nt) {
            v16h bf = *(const v16h*)(wout + (((size_t)kt * 4 + nt) * 32 + lane) * 16);
            acc0[nt] = __builtin_amdgcn_wmma_f32_16x16x32_f16(
                false, a0, false, bf, (short)0, acc0[nt], false, false);
            acc1[nt] = __builtin_amdgcn_wmma_f32_16x16x32_f16(
                false, a1, false, bf, (short)0, acc1[nt], false, false);
        }
    }
    const int n0 = lane & 15;
    const int rb = (lane < 16) ? 0 : 8;
    #pragma unroll
    for (int nt = 0; nt < 4; ++nt) {
        int n = nt * 16 + n0;
        float bias = out_b[n];
        #pragma unroll
        for (int j = 0; j < 8; ++j) {
            float va = fmaxf(acc0[nt][j] + bias, 0.f);
            float vb = fmaxf(acc1[nt][j] + bias, 0.f);
            h[((size_t)(b * VV + v0 + rb + j)) * 64 + n]      = (_Float16)va;
            h[((size_t)(b * VV + v0 + 16 + rb + j)) * 64 + n] = (_Float16)vb;
        }
    }
}

// =====================================================================
// 6) 3x3x3 wrap conv as implicit GEMM + bias + relu + residual.
//    wave = 32 output positions x 64 cout (two 16-row tiles reuse every
//    B fragment); dz unrolled x3 -> 48 static WMMAs, 432 issued per wave.
// =====================================================================
__global__ void conv_kernel(const _Float16* __restrict__ h,
                            const _Float16* __restrict__ wconv,
                            const float* __restrict__ conv_b,
                            const float* __restrict__ x,
                            float* __restrict__ outp) {
    const int lane = threadIdx.x & 31;
    const int wv   = threadIdx.x >> 5;
    const int b    = blockIdx.y;
    const int mtile = blockIdx.x * 8 + wv;     // 32-row tiles
    const int v0 = mtile * 32;
    const int v1 = v0 + 16;
    const int r = lane & 15;
    const int base = (lane < 16) ? 0 : 8;

    // tile 0 / tile 1 coordinates (tiles may sit in different z-lines)
    const int izA = v0 % 48 + r, iyA = (v0 / 48) % 48, ixA = v0 / 2304;
    const int izB = v1 % 48 + r, iyB = (v1 / 48) % 48, ixB = v1 / 2304;

    v8f acc0[4] = { vzero8(), vzero8(), vzero8(), vzero8() };
    v8f acc1[4] = { vzero8(), vzero8(), vzero8(), vzero8() };

    const _Float16* hb = h + (size_t)b * VV * 64;

    for (int txy = 0; txy < 9; ++txy) {
        const int dx = txy / 3 - 1, dy = txy % 3 - 1;
        const int qxA = wrap48(ixA + dx), qyA = wrap48(iyA + dy);
        const int qxB = wrap48(ixB + dx), qyB = wrap48(iyB + dy);
        const int rowA = (qxA * 48 + qyA) * 48;
        const int rowB = (qxB * 48 + qyB) * 48;
        #pragma unroll
        for (int dz = 0; dz < 3; ++dz) {
            const int t = txy * 3 + dz;
            const int qzA = wrap48(izA + dz - 1);
            const int qzB = wrap48(izB + dz - 1);
            const _Float16* arow0 = hb + ((size_t)(rowA + qzA)) * 64;
            const _Float16* arow1 = hb + ((size_t)(rowB + qzB)) * 64;
            #pragma unroll
            for (int kt = 0; kt < 2; ++kt) {
                int ks = kt * 32 + base;
                v16h a0 = cat8(*(const v8h*)(arow0 + ks), *(const v8h*)(arow0 + ks + 16));
                v16h a1 = cat8(*(const v8h*)(arow1 + ks), *(const v8h*)(arow1 + ks + 16));
                #pragma unroll
                for (int nt = 0; nt < 4; ++nt) {
                    v16h bf = *(const v16h*)(wconv +
                        ((((size_t)t * 2 + kt) * 4 + nt) * 32 + lane) * 16);
                    acc0[nt] = __builtin_amdgcn_wmma_f32_16x16x32_f16(
                        false, a0, false, bf, (short)0, acc0[nt], false, false);
                    acc1[nt] = __builtin_amdgcn_wmma_f32_16x16x32_f16(
                        false, a1, false, bf, (short)0, acc1[nt], false, false);
                }
            }
        }
    }
    const int n0 = lane & 15;
    const int rb = (lane < 16) ? 0 : 8;
    #pragma unroll
    for (int nt = 0; nt < 4; ++nt) {
        int n = nt * 16 + n0;
        float bias = conv_b[n];
        #pragma unroll
        for (int j = 0; j < 8; ++j) {
            size_t oiA = ((size_t)(b * CC + n)) * VV + (v0 + rb + j);
            size_t oiB = ((size_t)(b * CC + n)) * VV + (v1 + rb + j);
            outp[oiA] = fmaxf(acc0[nt][j] + bias, 0.f) + x[oiA];
            outp[oiB] = fmaxf(acc1[nt][j] + bias, 0.f) + x[oiB];
        }
    }
}

// =====================================================================
extern "C" void kernel_launch(void* const* d_in, const int* in_sizes, int n_in,
                              void* d_out, int out_size, void* d_ws, size_t ws_size,
                              hipStream_t stream) {
    const float* x          = (const float*)d_in[0];
    const float* mol_embed  = (const float*)d_in[1];
    const float* pos        = (const float*)d_in[2];
    const float* origin     = (const float*)d_in[3];
    const float* lattice    = (const float*)d_in[4];
    const int*   batch_nodes= (const int*)d_in[5];
    const float* v_w        = (const float*)d_in[6];
    const float* v_b        = (const float*)d_in[7];
    const float* out_w      = (const float*)d_in[8];
    const float* out_b      = (const float*)d_in[9];
    const float* conv_w     = (const float*)d_in[10];
    const float* conv_b     = (const float*)d_in[11];
    float* outp = (float*)d_out;

    _Float16* ws    = (_Float16*)d_ws;
    _Float16* vmat  = ws + OFF_VMAT;
    _Float16* vpack = ws + OFF_VPACK;
    _Float16* wout  = ws + OFF_WOUT;
    _Float16* wconv = ws + OFF_WCONV;
    _Float16* hin   = ws + OFF_HIN;
    _Float16* h     = ws + OFF_H;

    // 1) small projection v = mol_embed @ v_w.T + v_b
    prep_v_kernel<<<dim3((BB*AA*CC + 255) / 256), dim3(256), 0, stream>>>(
        mol_embed, v_w, v_b, vmat);

    // 2) pack all B matrices into WMMA fragment layout (needs vmat)
    pack_all_kernel<<<dim3((512 + 216*32 + BB*256 + 255) / 256), dim3(256), 0, stream>>>(
        out_w, conv_w, vmat, wout, wconv, vpack);

    // 3) transpose x to f16 row-major activations (first 64 cols of hin)
    prep_xt_kernel<<<dim3(BB * CC * (VV/4) / 256), dim3(256), 0, stream>>>(x, hin);

    // 4) attention + att@v (WMMA) -> cols 64..127 of hin
    attn_kernel<<<dim3(VV/16/4, BB), dim3(128), 0, stream>>>(
        pos, origin, lattice, batch_nodes, vpack, hin);

    // 5) out-projection GEMM + relu -> h  (32 rows per wave)
    out_gemm_kernel<<<dim3(VV/32/8, BB), dim3(256), 0, stream>>>(hin, wout, out_b, h);

    // 6) implicit-GEMM wrap conv + bias + relu + residual (32 rows per wave)
    conv_kernel<<<dim3(VV/32/8, BB), dim3(256), 0, stream>>>(h, wconv, conv_b, x, outp);
}
